// GraphTransformer_19713899888844
// MI455X (gfx1250) — compile-verified, matrix operand order checked by reading, not run on hardware
//
#include <hip/hip_runtime.h>
#include <hip/hip_bf16.h>
#include <math.h>

// ---------------------------------------------------------------------------
// Graph Transformer forward (B=1, N=256, DIM=256, DEPTH=2, H=8, Dh=64, E=16)
// Edge-projection refactor:
//   sim_ij = q_i.k_j + (We_h^T q_i).e_ij + q_i.be_h
//   out_i  = attn_i @ V + (sum_j attn_ij e_ij) @ We_h + be_h
// GEMMs via v_wmma_f32_16x16x32_f16, f16 operands pre-staged in workspace so
// every fragment load is a 16-byte b128 (no per-element cvt in inner loops).
// ---------------------------------------------------------------------------

typedef __attribute__((ext_vector_type(16))) _Float16 v16h;
typedef __attribute__((ext_vector_type(8)))  _Float16 v8h;
typedef __attribute__((ext_vector_type(8)))  float    v8f;

#define NN     256
#define DIMC   256
#define NHEADS 8
#define DH     64
#define INNERC 512
#define EDIM   16
#define NEGMAX 3.402823466e38f

// Load a 16-half A/B fragment (K-contiguous storage) as two b128 loads.
// Layout per ISA 7.12.2: lane half h owns K = k0+{8h..8h+7} and k0+{16+8h..}.
__device__ __forceinline__ v16h load_frag16(const _Float16* base, int k0,
                                            int half) {
  const v8h lo = *(const v8h*)(base + k0 + half * 8);
  const v8h hi = *(const v8h*)(base + k0 + 16 + half * 8);
  v16h f;
#pragma unroll
  for (int i = 0; i < 8; ++i) { f[i] = lo[i]; f[8 + i] = hi[i]; }
  return f;
}

// ---------------------------------------------------------------------------
// Weight convert + transpose: Wt[n, k] = (f16) W[k, n]
// ---------------------------------------------------------------------------
__global__ __launch_bounds__(256) void convt_kernel(
    const float* __restrict__ W, _Float16* __restrict__ Wt, int K, int Nout) {
  const int idx = blockIdx.x * 256 + threadIdx.x;
  const int k = idx / Nout, n = idx % Nout;
  Wt[n * K + k] = (_Float16)W[k * Nout + n];
}

// ---------------------------------------------------------------------------
// GEMM: C = act(A[M,K] @ Wt[N,K]^T + bias). One wave per 16x16 tile.
// ACT: 0 none, 1 exact GELU. Optional f32 / f16 / f16-transposed outputs.
// ---------------------------------------------------------------------------
template <int ACT, bool OUTF32, bool OUTF16, bool OUTF16T>
__global__ __launch_bounds__(32) void gemm16_wmma(
    const _Float16* __restrict__ A, const _Float16* __restrict__ Wt,
    const float* __restrict__ bias, float* __restrict__ C,
    _Float16* __restrict__ C16, _Float16* __restrict__ CT16,
    int M, int K, int Nout) {
  const int lane = threadIdx.x & 31;
  const int half = lane >> 4, nl = lane & 15;
  const int nt = blockIdx.x * 16, mt = blockIdx.y * 16;
  const _Float16* Ab = A + (size_t)(mt + nl) * K;
  const _Float16* Wb = Wt + (size_t)(nt + nl) * K;

  v8f acc = {};
  for (int k0 = 0; k0 < K; k0 += 32) {
    if (k0 + 32 < K) {  // next-iteration prefetch -> global_prefetch_b8
      __builtin_prefetch(Ab + k0 + 32, 0, 0);
      __builtin_prefetch(Wb + k0 + 32, 0, 0);
    }
    const v16h a = load_frag16(Ab, k0, half);
    const v16h b = load_frag16(Wb, k0, half);
    acc = __builtin_amdgcn_wmma_f32_16x16x32_f16(false, a, false, b,
                                                 (short)0, acc, false, false);
  }
#pragma unroll
  for (int r = 0; r < 8; ++r) {
    const int row = mt + r + 8 * half, col = nt + nl;
    float v = acc[r] + bias[col];
    if (ACT == 1) v = 0.5f * v * (1.0f + erff(v * 0.70710678118654752f));
    if (OUTF32)  C[(size_t)row * Nout + col] = v;
    if (OUTF16)  C16[(size_t)row * Nout + col] = (_Float16)v;
    if (OUTF16T) CT16[(size_t)col * M + row] = (_Float16)v;
  }
}

// ---------------------------------------------------------------------------
// Edge LayerNorm over last dim (16). One thread per (i,j) pair. f32 out.
// ---------------------------------------------------------------------------
__global__ __launch_bounds__(256) void edge_ln_kernel(
    const float* __restrict__ e, const float* __restrict__ g,
    const float* __restrict__ b, float* __restrict__ o) {
  const size_t p = (size_t)blockIdx.x * 256 + threadIdx.x;
  const float* ep = e + p * EDIM;
  float vals[EDIM], s = 0.f;
#pragma unroll
  for (int c = 0; c < EDIM; ++c) { vals[c] = ep[c]; s += vals[c]; }
  const float mean = s * (1.0f / EDIM);
  float v = 0.f;
#pragma unroll
  for (int c = 0; c < EDIM; ++c) { float d = vals[c] - mean; v += d * d; }
  const float r = rsqrtf(v * (1.0f / EDIM) + 1e-5f);
  float* op = o + p * EDIM;
#pragma unroll
  for (int c = 0; c < EDIM; ++c) op[c] = (vals[c] - mean) * r * g[c] + b[c];
}

// ---------------------------------------------------------------------------
// Row LayerNorm (dim 256), f32 in -> f16 out. One block per row.
// ---------------------------------------------------------------------------
__global__ __launch_bounds__(256) void layernorm16_kernel(
    const float* __restrict__ x, const float* __restrict__ g,
    const float* __restrict__ b, _Float16* __restrict__ y) {
  __shared__ float red[256];
  const int row = blockIdx.x, t = threadIdx.x;
  const float v = x[row * DIMC + t];
  red[t] = v;
  __syncthreads();
  for (int s = 128; s > 0; s >>= 1) {
    if (t < s) red[t] += red[t + s];
    __syncthreads();
  }
  const float mean = red[0] * (1.0f / DIMC);
  __syncthreads();
  const float d = v - mean;
  red[t] = d * d;
  __syncthreads();
  for (int s = 128; s > 0; s >>= 1) {
    if (t < s) red[t] += red[t + s];
    __syncthreads();
  }
  const float r = rsqrtf(red[0] * (1.0f / DIMC) + 1e-5f);
  y[row * DIMC + t] = (_Float16)(d * r * g[t] + b[t]);
}

// ---------------------------------------------------------------------------
// qe[i,h,c] = sum_d q[i,h,d] * We[c, h*64+d];  qbe[i,h] = q[i,h,:] . be[h,:]
// One thread per (i,h). q in f32 for precision.
// ---------------------------------------------------------------------------
__global__ __launch_bounds__(256) void qe_kernel(
    const float* __restrict__ q, const float* __restrict__ We,
    const float* __restrict__ be, float* __restrict__ qe,
    float* __restrict__ qbe) {
  const int t = blockIdx.x * 256 + threadIdx.x;
  const int i = t >> 3, h = t & 7;
  const float* qp = q + i * INNERC + h * DH;
  float acc[EDIM];
#pragma unroll
  for (int c = 0; c < EDIM; ++c) acc[c] = 0.f;
  float accb = 0.f;
  for (int d = 0; d < DH; ++d) {
    const float qd = qp[d];
    accb += qd * be[h * DH + d];
#pragma unroll
    for (int c = 0; c < EDIM; ++c) acc[c] += qd * We[c * INNERC + h * DH + d];
  }
#pragma unroll
  for (int c = 0; c < EDIM; ++c) qe[(i * NHEADS + h) * EDIM + c] = acc[c];
  qbe[i * NHEADS + h] = accb;
}

// ---------------------------------------------------------------------------
// Fused edge-biased attention. Block = 256 threads (8 waves).
// grid = (N/16 i-tiles, HEADS).
//   q16  [N, INNER]      row-contiguous d      (A frags, phase 1)
//   kv16 [N, 2*INNER]    row-contiguous d      (B frags = K^T, phase 1)
//   kvT16[2*INNER, N]    row-contiguous j      (B frags = V, phase 5)
// ---------------------------------------------------------------------------
__global__ __launch_bounds__(256) void attn_kernel(
    const _Float16* __restrict__ q16, const _Float16* __restrict__ kv16,
    const _Float16* __restrict__ kvT16, const float* __restrict__ enorm,
    const float* __restrict__ qe, const float* __restrict__ qbe,
    const float* __restrict__ We, const float* __restrict__ be,
    const unsigned char* __restrict__ mask, _Float16* __restrict__ out16) {
  __shared__ float s_sim[16][NN];
  __shared__ __align__(16) _Float16 s_prob[16][NN];
  __shared__ float s_qe[16][EDIM];
  __shared__ float s_qbe[16];
  __shared__ float s_ae[16][EDIM];

  const int tid = threadIdx.x;
  const int lane = tid & 31, wave = tid >> 5;
  const int half = lane >> 4, nl = lane & 15;
  const int i0 = blockIdx.x * 16;
  const int h = blockIdx.y;

  {  // stage qe / qbe
    const int i = tid >> 4, c = tid & 15;
    s_qe[i][c] = qe[((i0 + i) * NHEADS + h) * EDIM + c];
    if (c == 0) s_qbe[i] = qbe[(i0 + i) * NHEADS + h];
  }

  // -------- Phase 1: raw Q K^T via WMMA (2 j-tiles per wave) --------
  const _Float16* qrow = q16 + (size_t)(i0 + nl) * INNERC + h * DH;
  for (int jt = wave * 2; jt < wave * 2 + 2; ++jt) {
    const int j0 = jt * 16;
    const _Float16* krow = kv16 + (size_t)(j0 + nl) * (2 * INNERC) + h * DH;
    v8f acc = {};
#pragma unroll
    for (int k0 = 0; k0 < DH; k0 += 32) {
      const v16h a = load_frag16(qrow, k0, half);
      const v16h b = load_frag16(krow, k0, half);
      acc = __builtin_amdgcn_wmma_f32_16x16x32_f16(false, a, false, b,
                                                   (short)0, acc, false, false);
    }
#pragma unroll
    for (int r = 0; r < 8; ++r) s_sim[r + 8 * half][j0 + nl] = acc[r];
  }
  __syncthreads();

  // -------- Phase 2: edge bias (16-dim dot) + scale + mask --------
  {
    const int j = tid;
    const bool mj = mask[j] != 0;
    for (int i = 0; i < 16; ++i) {
      const float* ep = enorm + ((size_t)(i0 + i) * NN + j) * EDIM;
      __builtin_prefetch(ep + NN * EDIM, 0, 0);  // next row
      float dot = 0.f;
#pragma unroll
      for (int c = 0; c < EDIM; ++c) dot += s_qe[i][c] * ep[c];
      float v = (s_sim[i][j] + dot + s_qbe[i]) * 0.125f;  // 64^-0.5
      if (!(mj && mask[i0 + i] != 0)) v = -NEGMAX;
      s_sim[i][j] = v;
    }
  }
  __syncthreads();

  // -------- Phase 3: softmax, 2 rows per wave; probs -> f16 LDS --------
  for (int row = wave * 2; row < wave * 2 + 2; ++row) {
    float m = -NEGMAX;
    for (int j = lane; j < NN; j += 32) m = fmaxf(m, s_sim[row][j]);
#pragma unroll
    for (int off = 16; off > 0; off >>= 1) m = fmaxf(m, __shfl_xor(m, off, 32));
    float sum = 0.f;
    for (int j = lane; j < NN; j += 32) {
      const float e = __expf(s_sim[row][j] - m);
      s_sim[row][j] = e;
      sum += e;
    }
#pragma unroll
    for (int off = 16; off > 0; off >>= 1) sum += __shfl_xor(sum, off, 32);
    const float inv = 1.0f / sum;
    for (int j = lane; j < NN; j += 32)
      s_prob[row][j] = (_Float16)(s_sim[row][j] * inv);
  }
  __syncthreads();

  // -------- Phase 4: ae[i,c] = sum_j attn[i,j] * e[i,j,c] --------
  {
    const int i = tid >> 4, c = tid & 15;
    const float* ep = enorm + (size_t)(i0 + i) * NN * EDIM + c;
    float s = 0.f;
    for (int j = 0; j < NN; ++j) s += (float)s_prob[i][j] * ep[j * EDIM];
    s_ae[i][c] = s;
  }
  __syncthreads();

  // -------- Phase 5: out = attn @ V + ae @ We_h + be_h (4 waves) --------
  if (wave < 4) {
    const _Float16* vrow =
        kvT16 + (size_t)(INNERC + h * DH + wave * 16 + nl) * NN;
    const _Float16* prow = &s_prob[nl][0];
    v8f acc = {};
    for (int j0 = 0; j0 < NN; j0 += 32) {
      const v16h a = load_frag16(prow, j0, half);
      const v16h b = load_frag16(vrow, j0, half);
      acc = __builtin_amdgcn_wmma_f32_16x16x32_f16(false, a, false, b,
                                                   (short)0, acc, false, false);
    }
    const int col = h * DH + wave * 16 + nl;
#pragma unroll
    for (int r = 0; r < 8; ++r) {
      const int m = r + 8 * half;
      float val = acc[r] + be[col];  // sum_j attn = 1
#pragma unroll
      for (int c = 0; c < EDIM; ++c) val += s_ae[m][c] * We[c * INNERC + col];
      out16[(size_t)(i0 + m) * INNERC + col] = (_Float16)val;
    }
  }
}

// ---------------------------------------------------------------------------
// Gated residual: gate = sigmoid([x,res,x-res] @ Wg); out = x*g + res*(1-g)
// One block per row. Safe in-place (out may alias res).
// ---------------------------------------------------------------------------
__global__ __launch_bounds__(256) void gated_residual_kernel(
    const float* __restrict__ x, const float* __restrict__ res,
    const float* __restrict__ Wg, float* __restrict__ out) {
  __shared__ float red[256];
  const int row = blockIdx.x, t = threadIdx.x;
  const float xv = x[row * DIMC + t], rv = res[row * DIMC + t];
  red[t] = xv * Wg[t] + rv * Wg[DIMC + t] + (xv - rv) * Wg[2 * DIMC + t];
  __syncthreads();
  for (int s = 128; s > 0; s >>= 1) {
    if (t < s) red[t] += red[t + s];
    __syncthreads();
  }
  const float gate = 1.0f / (1.0f + __expf(-red[0]));
  out[row * DIMC + t] = xv * gate + rv * (1.0f - gate);
}

// ---------------------------------------------------------------------------
// Host orchestration
// Input order (setup_inputs insertion order, params flattened in order):
//   0 nodes, 1 edges, 2 mask, 3 edge_ln_g, 4 edge_ln_b,
//   then per layer (18): attn_ln_g, attn_ln_b, Wq, bq, Wkv, bkv, We, be,
//   Wo, bo, Wg_attn, ff_ln_g, ff_ln_b, W1, b1, W2, b2, Wg_ff
// ---------------------------------------------------------------------------
extern "C" void kernel_launch(void* const* d_in, const int* in_sizes, int n_in,
                              void* d_out, int out_size, void* d_ws,
                              size_t ws_size, hipStream_t stream) {
  (void)in_sizes; (void)n_in; (void)out_size; (void)ws_size;
  const float* nodes = (const float*)d_in[0];
  const float* edges = (const float*)d_in[1];
  const unsigned char* mask = (const unsigned char*)d_in[2];
  const float* edge_ln_g = (const float*)d_in[3];
  const float* edge_ln_b = (const float*)d_in[4];
  auto P = [&](int layer, int field) -> const float* {
    return (const float*)d_in[5 + layer * 18 + field];
  };
  enum { ATTN_LN_G, ATTN_LN_B, WQ, BQ, WKV, BKV, WE, BE, WO, BO, WG_ATTN,
         FF_LN_G, FF_LN_B, W1, B1, W2, B2, WG_FF };

  // ---- workspace carve-up ----
  float* fws = (float*)d_ws;
  float* e_norm    = fws; fws += (size_t)NN * NN * EDIM;  // 1,048,576
  float* nodes_cur = fws; fws += NN * DIMC;
  float* q_f32     = fws; fws += NN * INNERC;
  float* qe        = fws; fws += NN * NHEADS * EDIM;
  float* qbe       = fws; fws += NN * NHEADS;
  float* proj      = fws; fws += NN * DIMC;
  float* yb        = fws; fws += NN * DIMC;

  _Float16* hws = (_Float16*)fws;
  _Float16* x16    = hws; hws += NN * DIMC;
  _Float16* q16    = hws; hws += NN * INNERC;
  _Float16* kv16   = hws; hws += NN * 2 * INNERC;
  _Float16* kvT16  = hws; hws += NN * 2 * INNERC;
  _Float16* attn16 = hws; hws += NN * INNERC;
  _Float16* ffh16  = hws; hws += NN * DIMC * 4;
  _Float16* wqt    = hws; hws += DIMC * INNERC;
  _Float16* wkvt   = hws; hws += DIMC * 2 * INNERC;
  _Float16* wot    = hws; hws += INNERC * DIMC;
  _Float16* w1t    = hws; hws += DIMC * DIMC * 4;
  _Float16* w2t    = hws; hws += DIMC * 4 * DIMC;

  hipMemcpyAsync(nodes_cur, nodes, sizeof(float) * NN * DIMC,
                 hipMemcpyDeviceToDevice, stream);
  edge_ln_kernel<<<NN * NN / 256, 256, 0, stream>>>(edges, edge_ln_g,
                                                    edge_ln_b, e_norm);

  for (int l = 0; l < 2; ++l) {
    // ---- stage this layer's weights as f16 [Nout, K] ----
    convt_kernel<<<DIMC * INNERC / 256, 256, 0, stream>>>(P(l, WQ), wqt,
                                                          DIMC, INNERC);
    convt_kernel<<<DIMC * 2 * INNERC / 256, 256, 0, stream>>>(
        P(l, WKV), wkvt, DIMC, 2 * INNERC);
    convt_kernel<<<INNERC * DIMC / 256, 256, 0, stream>>>(P(l, WO), wot,
                                                          INNERC, DIMC);
    convt_kernel<<<DIMC * DIMC * 4 / 256, 256, 0, stream>>>(P(l, W1), w1t,
                                                            DIMC, DIMC * 4);
    convt_kernel<<<DIMC * 4 * DIMC / 256, 256, 0, stream>>>(P(l, W2), w2t,
                                                            DIMC * 4, DIMC);

    // ---- attention branch ----
    layernorm16_kernel<<<NN, 256, 0, stream>>>(nodes_cur, P(l, ATTN_LN_G),
                                               P(l, ATTN_LN_B), x16);
    gemm16_wmma<0, true, true, false>
        <<<dim3(INNERC / 16, NN / 16), 32, 0, stream>>>(
            x16, wqt, P(l, BQ), q_f32, q16, nullptr, NN, DIMC, INNERC);
    gemm16_wmma<0, false, true, true>
        <<<dim3(2 * INNERC / 16, NN / 16), 32, 0, stream>>>(
            x16, wkvt, P(l, BKV), nullptr, kv16, kvT16, NN, DIMC, 2 * INNERC);
    qe_kernel<<<NN * NHEADS / 256, 256, 0, stream>>>(q_f32, P(l, WE),
                                                     P(l, BE), qe, qbe);
    attn_kernel<<<dim3(NN / 16, NHEADS), 256, 0, stream>>>(
        q16, kv16, kvT16, e_norm, qe, qbe, P(l, WE), P(l, BE), mask, attn16);
    gemm16_wmma<0, true, false, false>
        <<<dim3(DIMC / 16, NN / 16), 32, 0, stream>>>(
            attn16, wot, P(l, BO), proj, nullptr, nullptr, NN, INNERC, DIMC);
    gated_residual_kernel<<<NN, 256, 0, stream>>>(proj, nodes_cur,
                                                  P(l, WG_ATTN), nodes_cur);

    // ---- feedforward branch ----
    layernorm16_kernel<<<NN, 256, 0, stream>>>(nodes_cur, P(l, FF_LN_G),
                                               P(l, FF_LN_B), x16);
    gemm16_wmma<1, false, true, false>
        <<<dim3(DIMC * 4 / 16, NN / 16), 32, 0, stream>>>(
            x16, w1t, P(l, B1), nullptr, ffh16, nullptr, NN, DIMC, DIMC * 4);
    gemm16_wmma<0, true, false, false>
        <<<dim3(DIMC / 16, NN / 16), 32, 0, stream>>>(
            ffh16, w2t, P(l, B2), yb, nullptr, nullptr, NN, DIMC * 4, DIMC);
    float* outp = (l == 1) ? (float*)d_out : nodes_cur;
    gated_residual_kernel<<<NN, 256, 0, stream>>>(yb, nodes_cur, P(l, WG_FF),
                                                  outp);
  }
}
